// AttentionBlock_67680094650946
// MI455X (gfx1250) — matrix operand order repaired and checked by reading, not compiled, not run on hardware
//
#include <hip/hip_runtime.h>

typedef __attribute__((ext_vector_type(16))) _Float16 v16h;
typedef __attribute__((ext_vector_type(8)))  _Float16 v8h;
typedef __attribute__((ext_vector_type(4)))  _Float16 v4h;
typedef __attribute__((ext_vector_type(8)))  float    v8f;

#define DIMS 1024
#define KSZ  128
#define SEQ  2048
#define NB   4
#define ROWS (NB * SEQ)      // 8192
#define OUTD (DIMS + KSZ)    // 1152

__device__ __forceinline__ v8f wmma_f16(v16h a, v16h b, v8f c) {
  // (neg_a, A, neg_b, B, c_mod, C, reuse_a, reuse_b)
  return __builtin_amdgcn_wmma_f32_16x16x32_f16(false, a, false, b, (short)0, c,
                                                false, false);
}

// --- Stage 1: X fp32 -> f16, and write concat pass-through columns of out ---
__global__ __launch_bounds__(256) void cvt_x_kernel(const float* __restrict__ X,
                                                    _Float16* __restrict__ Xh,
                                                    float* __restrict__ out) {
  size_t t = (size_t)blockIdx.x * blockDim.x + threadIdx.x;
  size_t base = t * 4;  // element index, exact coverage of ROWS*DIMS
  float4 xv = *(const float4*)(X + base);
  v4h h = {(_Float16)xv.x, (_Float16)xv.y, (_Float16)xv.z, (_Float16)xv.w};
  *(v4h*)(Xh + base) = h;
  size_t row = base >> 10;      // /1024 (4 elems never cross a row)
  size_t col = base & 1023;
  *(float4*)(out + row * OUTD + col) = xv;
}

// --- Stage 1b: W fp32 -> f16 transposed: WhT[mat][n*1024 + d] = W[d*128+n] ---
__global__ __launch_bounds__(256) void cvt_w_kernel(const float* __restrict__ Wq,
                                                    const float* __restrict__ Wk,
                                                    const float* __restrict__ Wv,
                                                    _Float16* __restrict__ WhT) {
  int idx = blockIdx.x * 256 + threadIdx.x;
  if (idx >= 3 * DIMS * KSZ) return;
  int mat = idx / (DIMS * KSZ);
  int rem = idx - mat * (DIMS * KSZ);
  int d = rem >> 7, nn = rem & 127;
  const float* src = (mat == 0) ? Wq : (mat == 1) ? Wk : Wv;
  WhT[(size_t)mat * DIMS * KSZ + (size_t)nn * DIMS + d] = (_Float16)src[rem];
}

// --- Stage 2: QKV projection. grid=(ROWS/16, 3), block=256 (8 waves). ---
// Wave w computes the 16x16 output tile at N-tile w via 32 WMMA K-steps.
__global__ __launch_bounds__(256) void qkv_kernel(
    const _Float16* __restrict__ Xh, const _Float16* __restrict__ WhT,
    const float* __restrict__ bq, const float* __restrict__ bk,
    const float* __restrict__ bv, _Float16* __restrict__ qh,
    _Float16* __restrict__ kh, _Float16* __restrict__ vhT) {
  const int wave = threadIdx.x >> 5;
  const int lane = threadIdx.x & 31;
  const int l16 = lane & 15, half = lane >> 4;
  const int row0 = blockIdx.x * 16;
  const int mat = blockIdx.y;
  const int n0 = wave * 16;
  const _Float16* Wt =
      WhT + (size_t)mat * DIMS * KSZ + (size_t)(n0 + l16) * DIMS;
  const _Float16* Xrow = Xh + (size_t)(row0 + l16) * DIMS;
  v8f c = {};
  for (int k0 = 0; k0 < DIMS; k0 += 32) {
    // A fragment (16x32 f16): lane=row, contiguous 16 halfs along K at half*16
    v8h alo = *(const v8h*)(Xrow + k0 + half * 16);
    v8h ahi = *(const v8h*)(Xrow + k0 + half * 16 + 8);
    v16h a;
#pragma unroll
    for (int j = 0; j < 8; ++j) { a[j] = alo[j]; a[j + 8] = ahi[j]; }
    // B fragment (32x16 f16): lane=col, contiguous 16 halfs along K (W is transposed)
    v16h b = *(const v16h*)(Wt + k0 + half * 16);
    c = wmma_f16(a, b, c);
  }
  const float* bias = (mat == 0) ? bq : (mat == 1) ? bk : bv;
  float bb = bias[n0 + l16];
#pragma unroll
  for (int r = 0; r < 8; ++r) {
    int row = row0 + r + half * 8;   // C layout: reg r -> row r + half*8
    int col = n0 + l16;
    float v = c[r] + bb;
    if (mat == 0) {
      qh[(size_t)row * KSZ + col] = (_Float16)v;
    } else if (mat == 1) {
      kh[(size_t)row * KSZ + col] = (_Float16)v;
    } else {  // store V transposed per batch: [b][col][seq]
      int bs = row >> 11, s = row & (SEQ - 1);
      vhT[((size_t)bs * KSZ + col) * SEQ + s] = (_Float16)v;
    }
  }
}

// --- Stage 3: causal flash attention. grid=(SEQ/16, NB), block=32 (1 wave). ---
__global__ __launch_bounds__(32) void attn_kernel(
    const _Float16* __restrict__ qh, const _Float16* __restrict__ kh,
    const _Float16* __restrict__ vhT, float* __restrict__ out) {
  __shared__ __align__(32) _Float16 lds_p[16 * 32];  // P tile, C->A transpose
  const int lane = threadIdx.x;
  const int n = lane & 15, half = lane >> 4;
  const int q0 = blockIdx.x * 16;
  const int b = blockIdx.y;
  const size_t rowbase = (size_t)b * SEQ;

  // Preload Q A-fragments (16x128 = 4 fragments along K)
  v16h qa[4];
#pragma unroll
  for (int kk = 0; kk < 4; ++kk) {
    const _Float16* ap = qh + (rowbase + q0 + n) * KSZ + kk * 32 + half * 16;
    v8h lo = *(const v8h*)ap;
    v8h hi = *(const v8h*)(ap + 8);
#pragma unroll
    for (int j = 0; j < 8; ++j) { qa[kk][j] = lo[j]; qa[kk][j + 8] = hi[j]; }
  }

  v8f o[8] = {};           // O accumulator: 8 N-tiles x (reg r -> row r+half*8)
  float mrun[8], lrun[8];
#pragma unroll
  for (int r = 0; r < 8; ++r) { mrun[r] = -3.0e38f; lrun[r] = 0.0f; }
  const float scale = 0.08838834764831845f;  // 1/sqrt(128)

  for (int j0 = 0; j0 < q0 + 16; j0 += 32) {  // 32 keys per step, causal bound
    v8f sL = {}, sH = {};
#pragma unroll
    for (int kk = 0; kk < 4; ++kk) {
      // B fragment = K^T: lane=key col, contiguous along head-dim
      v16h bL =
          *(const v16h*)(kh + (rowbase + j0 + n) * KSZ + kk * 32 + half * 16);
      sL = wmma_f16(qa[kk], bL, sL);
      v16h bH = *(const v16h*)(kh + (rowbase + j0 + 16 + n) * KSZ + kk * 32 +
                               half * 16);
      sH = wmma_f16(qa[kk], bH, sH);
    }
    // Online softmax over 32 columns (row = reg r + half*8, col = lane%16)
    float eLs[8], eHs[8];
#pragma unroll
    for (int r = 0; r < 8; ++r) {
      const int qrow = q0 + r + half * 8;
      float vL = (j0 + n <= qrow) ? sL[r] * scale : -3.0e38f;
      float vH = (j0 + 16 + n <= qrow) ? sH[r] * scale : -3.0e38f;
      float tmax = fmaxf(vL, vH);
      tmax = fmaxf(tmax, __shfl_xor(tmax, 1, 32));
      tmax = fmaxf(tmax, __shfl_xor(tmax, 2, 32));
      tmax = fmaxf(tmax, __shfl_xor(tmax, 4, 32));
      tmax = fmaxf(tmax, __shfl_xor(tmax, 8, 32));
      float mnew = fmaxf(mrun[r], tmax);
      float alpha = __expf(mrun[r] - mnew);
      mrun[r] = mnew;
      float eL = __expf(vL - mnew);
      float eH = __expf(vH - mnew);
      float tsum = eL + eH;
      tsum += __shfl_xor(tsum, 1, 32);
      tsum += __shfl_xor(tsum, 2, 32);
      tsum += __shfl_xor(tsum, 4, 32);
      tsum += __shfl_xor(tsum, 8, 32);
      lrun[r] = lrun[r] * alpha + tsum;
      eLs[r] = eL; eHs[r] = eH;
#pragma unroll
      for (int t = 0; t < 8; ++t) o[t][r] *= alpha;
    }
    // Transpose P from C layout to A layout via LDS
    __syncthreads();
#pragma unroll
    for (int r = 0; r < 8; ++r) {
      int row = r + half * 8;
      lds_p[row * 32 + n] = (_Float16)eLs[r];
      lds_p[row * 32 + 16 + n] = (_Float16)eHs[r];
    }
    __syncthreads();
    v16h pa;
    {
      const _Float16* ap = lds_p + n * 32 + half * 16;  // lane = row m
      v8h lo = *(const v8h*)ap;
      v8h hi = *(const v8h*)(ap + 8);
#pragma unroll
      for (int j = 0; j < 8; ++j) { pa[j] = lo[j]; pa[j + 8] = hi[j]; }
    }
    // PV: 8 WMMAs, B from transposed V (contiguous along seq)
#pragma unroll
    for (int t = 0; t < 8; ++t) {
      v16h vb = *(const v16h*)(vhT + ((size_t)b * KSZ + t * 16 + n) * SEQ + j0 +
                               half * 16);
      o[t] = wmma_f16(pa, vb, o[t]);
    }
  }
  // Normalize and write the last 128 output columns
#pragma unroll
  for (int r = 0; r < 8; ++r) {
    const int qrow = q0 + r + half * 8;
    float inv = 1.0f / lrun[r];
    size_t ob = ((size_t)b * SEQ + qrow) * (size_t)OUTD + DIMS;
#pragma unroll
    for (int t = 0; t < 8; ++t) out[ob + t * 16 + n] = o[t][r] * inv;
  }
}

extern "C" void kernel_launch(void* const* d_in, const int* in_sizes, int n_in,
                              void* d_out, int out_size, void* d_ws,
                              size_t ws_size, hipStream_t stream) {
  const float* X  = (const float*)d_in[0];
  const float* Wq = (const float*)d_in[1];
  const float* bq = (const float*)d_in[2];
  const float* Wk = (const float*)d_in[3];
  const float* bk = (const float*)d_in[4];
  const float* Wv = (const float*)d_in[5];
  const float* bv = (const float*)d_in[6];
  float* out = (float*)d_out;

  // Workspace layout (~23.9 MB): Xh | WhT | qh | kh | vhT
  char* ws = (char*)d_ws;
  _Float16* Xh  = (_Float16*)ws;
  _Float16* WhT = (_Float16*)(ws + (size_t)ROWS * DIMS * 2);
  _Float16* qh  = (_Float16*)((char*)WhT + (size_t)3 * DIMS * KSZ * 2);
  _Float16* kh  = qh + (size_t)ROWS * KSZ;
  _Float16* vhT = kh + (size_t)ROWS * KSZ;

  cvt_x_kernel<<<ROWS * DIMS / 4 / 256, 256, 0, stream>>>(X, Xh, out);
  cvt_w_kernel<<<(3 * DIMS * KSZ + 255) / 256, 256, 0, stream>>>(Wq, Wk, Wv,
                                                                 WhT);
  qkv_kernel<<<dim3(ROWS / 16, 3), 256, 0, stream>>>(Xh, WhT, bq, bk, bv, qh,
                                                     kh, vhT);
  attn_kernel<<<dim3(SEQ / 16, NB), 32, 0, stream>>>(qh, kh, vhT, out);
}
